// Embed2GraphByAttention_53420803228027
// MI455X (gfx1250) — matrix operand order repaired and checked by reading, not compiled, not run on hardware
//
#include <hip/hip_runtime.h>
#include <hip/hip_bf16.h>

#define B_ 4
#define N_ 4096
#define D_ 128
#define TOPK_ 20

typedef __attribute__((ext_vector_type(16))) __bf16 v16bf;
typedef __attribute__((ext_vector_type(8)))  __bf16 v8bf;
typedef __attribute__((ext_vector_type(8)))  float  v8f;

__device__ __forceinline__ __bf16 f2bf(float f) {
    unsigned u = __builtin_bit_cast(unsigned, f);
    unsigned r = (u + 0x7FFFu + ((u >> 16) & 1u)) >> 16;
    return __builtin_bit_cast(__bf16, (unsigned short)r);
}

__device__ __forceinline__ float wred_max(float v) {
#pragma unroll
    for (int m = 16; m > 0; m >>= 1) v = fmaxf(v, __shfl_xor(v, m, 32));
    return v;
}
__device__ __forceinline__ float wred_min(float v) {
#pragma unroll
    for (int m = 16; m > 0; m >>= 1) v = fminf(v, __shfl_xor(v, m, 32));
    return v;
}
__device__ __forceinline__ float wred_sum(float v) {
#pragma unroll
    for (int m = 16; m > 0; m >>= 1) v += __shfl_xor(v, m, 32);
    return v;
}
__device__ __forceinline__ int wred_sumi(int v) {
#pragma unroll
    for (int m = 16; m > 0; m >>= 1) v += __shfl_xor(v, m, 32);
    return v;
}

// ---------------------------------------------------------------------------
// Kernel 0: transpose + convert weights to bf16.  Wt[e][d] = W[d][e]
// ---------------------------------------------------------------------------
__global__ void convert_w(const float* __restrict__ Wq, const float* __restrict__ Wk,
                          __bf16* __restrict__ Wqt, __bf16* __restrict__ Wkt) {
    const int e = blockIdx.x;
    const int d = threadIdx.x;
    const float* W = blockIdx.y ? Wk : Wq;
    __bf16* Wt = blockIdx.y ? Wkt : Wqt;
    Wt[e * D_ + d] = f2bf(W[d * D_ + e]);
}

// ---------------------------------------------------------------------------
// Kernel 1: Q = x*Wq + bq, Kp = x*Wk + bk, output bf16.
// One wave per 16-row tile; blockIdx.y selects Q (0) vs K (1).
// ---------------------------------------------------------------------------
__global__ __launch_bounds__(32) void proj_wmma(
    const float* __restrict__ x,
    const __bf16* __restrict__ Wqt, const float* __restrict__ bq,
    const __bf16* __restrict__ Wkt, const float* __restrict__ bk,
    __bf16* __restrict__ Qout, __bf16* __restrict__ Kout) {

    const int lane  = threadIdx.x & 31;
    const int tile  = blockIdx.x;            // 0 .. B*N/16 - 1
    const bool doK  = blockIdx.y != 0;
    const __bf16* Wt  = doK ? Wkt : Wqt;
    const float* bias = doK ? bk : bq;
    __bf16* out       = doK ? Kout : Qout;

    const int arow  = lane & 15;
    const int khalf = lane >> 4;             // 0 or 1
    const int row   = tile * 16 + arow;

    // A tiles (16x32 bf16 each), K = 0..127 in 4 steps.
    // Lane layout: lane<16 -> K = k0..k0+15, k0 = kk*32 (+16 for upper half)
    v16bf a[4];
#pragma unroll
    for (int kk = 0; kk < 4; ++kk) {
        const float* p = x + (size_t)row * D_ + kk * 32 + khalf * 16;
        float4 f0 = *(const float4*)(p);
        float4 f1 = *(const float4*)(p + 4);
        float4 f2 = *(const float4*)(p + 8);
        float4 f3 = *(const float4*)(p + 12);
        v16bf t;
        t[0] = f2bf(f0.x);  t[1] = f2bf(f0.y);  t[2] = f2bf(f0.z);  t[3] = f2bf(f0.w);
        t[4] = f2bf(f1.x);  t[5] = f2bf(f1.y);  t[6] = f2bf(f1.z);  t[7] = f2bf(f1.w);
        t[8] = f2bf(f2.x);  t[9] = f2bf(f2.y);  t[10] = f2bf(f2.z); t[11] = f2bf(f2.w);
        t[12] = f2bf(f3.x); t[13] = f2bf(f3.y); t[14] = f2bf(f3.z); t[15] = f2bf(f3.w);
        a[kk] = t;
    }

#pragma unroll
    for (int et = 0; et < 8; ++et) {
        const int e = et * 16 + arow;        // output column for this lane
        v8f c = {};
#pragma unroll
        for (int kk = 0; kk < 4; ++kk) {
            // B lane layout: col = lane%16, 16 consecutive K at kk*32 + (lane/16)*16
            v16bf bm = *(const v16bf*)(Wt + (size_t)e * D_ + kk * 32 + khalf * 16);
            c = __builtin_amdgcn_wmma_f32_16x16x32_bf16(false, a[kk], false, bm,
                                                        (short)0, c, false, false);
        }
        const float bv = bias[e];
#pragma unroll
        for (int r = 0; r < 8; ++r) {
            const int orow = tile * 16 + r + 8 * khalf;   // C layout: M = r + 8*(lane/16)
            out[(size_t)orow * D_ + e] = f2bf(c[r] + bv);
        }
    }
}

// ---------------------------------------------------------------------------
// Kernel 2: scores = Q*K^T/sqrt(D) for a 16-row tile (kept in LDS, 256 KB),
// then per-row top-20 threshold (wave bisection) + masked softmax -> d_out.
// Block = 256 threads (8 waves) per (row-tile, batch).
// ---------------------------------------------------------------------------
__global__ __launch_bounds__(256) void attn_topk_softmax(
    const __bf16* __restrict__ Qb, const __bf16* __restrict__ Kb,
    float* __restrict__ out) {

    extern __shared__ float s_scores[];      // [16][N_] = 256 KB
    const int lane  = threadIdx.x & 31;
    const int wv    = threadIdx.x >> 5;      // 0..7
    const int rt    = blockIdx.x;            // 0..N/16-1
    const int b     = blockIdx.y;            // 0..B-1
    const int arow  = lane & 15;
    const int khalf = lane >> 4;

    const __bf16* Qrow  = Qb + ((size_t)b * N_ + (size_t)rt * 16) * D_;
    const __bf16* Kbase = Kb + (size_t)b * N_ * D_;

    // Load A tiles (this tile's 16 Q rows), shared by all column tiles.
    v16bf a[4];
#pragma unroll
    for (int kk = 0; kk < 4; ++kk) {
        const __bf16* p = Qrow + (size_t)arow * D_ + kk * 32 + khalf * 16;
        v8bf c0 = *(const v8bf*)(p);
        v8bf c1 = *(const v8bf*)(p + 8);
        v16bf t;
#pragma unroll
        for (int i = 0; i < 8; ++i) { t[i] = c0[i]; t[8 + i] = c1[i]; }
        a[kk] = t;
    }

    const float scale = 0.08838834764831843f;   // 1/sqrt(128)

    // Score GEMM: 8 waves split 256 column tiles; 4 WMMAs per tile.
    for (int nt = wv; nt < N_ / 16; nt += 8) {
        const int m = nt * 16 + arow;
        const __bf16* kp = Kbase + (size_t)m * D_ + khalf * 16;
        if (nt + 8 < N_ / 16)
            __builtin_prefetch(Kbase + (size_t)(m + 128) * D_, 0, 1);
        v8f c = {};
#pragma unroll
        for (int kk = 0; kk < 4; ++kk) {
            v16bf bm = *(const v16bf*)(kp + kk * 32);
            c = __builtin_amdgcn_wmma_f32_16x16x32_bf16(false, a[kk], false, bm,
                                                        (short)0, c, false, false);
        }
#pragma unroll
        for (int r = 0; r < 8; ++r)
            s_scores[(size_t)(r + 8 * khalf) * N_ + nt * 16 + arow] = c[r] * scale;
    }
    __syncthreads();

    // Top-k + softmax: 2 rows per wave, 128 values per lane held in VGPRs.
    for (int rr = 0; rr < 2; ++rr) {
        const int row = wv * 2 + rr;
        float v[128];
#pragma unroll
        for (int i = 0; i < 128; ++i) v[i] = s_scores[(size_t)row * N_ + lane + 32 * i];

        float rmax = v[0], rmin = v[0];
#pragma unroll
        for (int i = 1; i < 128; ++i) { rmax = fmaxf(rmax, v[i]); rmin = fminf(rmin, v[i]); }
        rmax = wred_max(rmax);
        rmin = wred_min(rmin);

        // bisection: largest t with count(s >= t) >= TOPK
        float lo = rmin - 1.0f, hi = rmax + 1.0f;
        for (int it = 0; it < 30; ++it) {
            const float mid = 0.5f * (lo + hi);
            int c = 0;
#pragma unroll
            for (int i = 0; i < 128; ++i) c += (v[i] >= mid) ? 1 : 0;
            c = wred_sumi(c);
            if (c >= TOPK_) lo = mid; else hi = mid;
        }
        const float thr = lo;

        // softmax over sparse row: masked-out entries behave as value 0.
        const float m0 = fmaxf(rmax, 0.0f);
        int   cnt = 0;
        float se  = 0.0f;
#pragma unroll
        for (int i = 0; i < 128; ++i) {
            const bool keep = v[i] >= thr;
            cnt += keep ? 1 : 0;
            se  += keep ? __expf(v[i] - m0) : 0.0f;
        }
        cnt = wred_sumi(cnt);
        se  = wred_sum(se);
        const float e0   = __expf(-m0);                     // exp(0 - m0)
        const float Z    = se + (float)(N_ - cnt) * e0;
        const float invZ = 1.0f / Z;
        const float zoff = e0 * invZ;                       // output for non-topk

        float* orow = out + ((size_t)b * N_ + (size_t)rt * 16 + row) * (size_t)N_;
#pragma unroll
        for (int i = 0; i < 128; ++i) {
            const float val = (v[i] >= thr) ? __expf(v[i] - m0) * invZ : zoff;
            orow[lane + 32 * i] = val;
        }
    }
}

// ---------------------------------------------------------------------------
extern "C" void kernel_launch(void* const* d_in, const int* in_sizes, int n_in,
                              void* d_out, int out_size, void* d_ws, size_t ws_size,
                              hipStream_t stream) {
    const float* x  = (const float*)d_in[0];
    const float* Wq = (const float*)d_in[1];
    const float* bq = (const float*)d_in[2];
    const float* Wk = (const float*)d_in[3];
    const float* bk = (const float*)d_in[4];
    float* out = (float*)d_out;

    // Workspace layout
    char* ws = (char*)d_ws;
    const size_t qbytes = (size_t)B_ * N_ * D_ * sizeof(__bf16);   // 4 MB
    __bf16* Qbf = (__bf16*)(ws);
    __bf16* Kbf = (__bf16*)(ws + qbytes);
    __bf16* Wqt = (__bf16*)(ws + 2 * qbytes);
    __bf16* Wkt = (__bf16*)(ws + 2 * qbytes + (size_t)D_ * D_ * sizeof(__bf16));

    // 0) weights -> transposed bf16
    convert_w<<<dim3(D_, 2), D_, 0, stream>>>(Wq, Wk, Wqt, Wkt);

    // 1) projections (one wave per 16-row tile; y: 0=Q, 1=K)
    proj_wmma<<<dim3(B_ * N_ / 16, 2), 32, 0, stream>>>(x, Wqt, bq, Wkt, bk, Qbf, Kbf);

    // 2) scores + top-k + softmax; 256 KB dynamic LDS (16 rows x 4096 f32)
    const size_t smem = (size_t)16 * N_ * sizeof(float);
    attn_topk_softmax<<<dim3(N_ / 16, B_), 256, smem, stream>>>(Qbf, Kbf, out);
}
